// GAE_39127152066565
// MI455X (gfx1250) — matrix operand order repaired and checked by reading, not compiled, not run on hardware
//
#include <hip/hip_runtime.h>

// ---------- types ----------
typedef __attribute__((ext_vector_type(16))) __bf16 v16bf;
typedef __attribute__((ext_vector_type(8)))  __bf16 v8bf;
typedef __attribute__((ext_vector_type(8)))  float  v8f;
typedef __attribute__((ext_vector_type(4)))  float  v4f;

union FragU { v16bf v; v8bf h[2]; };

#define GAE_KC 128  // K-chunk staged in LDS per block iteration

// Use the Tensor Data Mover when the builtin exists (both probed toolchains).
#ifndef GAE_USE_TDM
#if __has_builtin(__builtin_amdgcn_tensor_load_to_lds)
#define GAE_USE_TDM 1
#else
#define GAE_USE_TDM 0
#endif
#endif

#if GAE_USE_TDM
#pragma message("GAE probe: TDM path ENABLED")
#if defined(__clang_major__) && (__clang_major__ >= 23)
#pragma message("GAE probe: using 6-arg tensor_load_to_lds (clang>=23)")
#else
#pragma message("GAE probe: using 5-arg tensor_load_to_lds (clang<23)")
#endif
#else
#pragma message("GAE probe: TDM path DISABLED (cooperative-copy fallback)")
#endif

#if GAE_USE_TDM
typedef __attribute__((ext_vector_type(4))) unsigned int u32x4;
typedef __attribute__((ext_vector_type(4))) int          i32x4;
typedef __attribute__((ext_vector_type(8))) int          i32x8;

// Issue one TDM 2D tile load: gsrc[tile_rows][tile_k] (row stride = stride_k
// elements, 2-byte elements) -> LDS[lds_off], rows contiguous (tile_k each).
// D# packing per CDNA5 ISA ch.8 (group0 128b, group1 256b; groups 2/3 zero).
__device__ __forceinline__ void gae_tdm_load_2d(const __bf16* gsrc,
                                                unsigned lds_off,
                                                unsigned tile_k,
                                                unsigned tile_rows,
                                                unsigned tensor_k,
                                                unsigned tensor_rows,
                                                unsigned stride_k) {
  unsigned long long ga = (unsigned long long)(size_t)gsrc;
  u32x4 g0;
  g0[0] = 1u;                                   // count=1 valid user descriptor
  g0[1] = lds_off;                              // lds_addr (bytes)
  g0[2] = (unsigned)(ga & 0xffffffffu);         // global_addr[31:0]
  g0[3] = (unsigned)((ga >> 32) & 0x1ffffffu)   // global_addr[56:32]
          | 0x80000000u;                        // type=2 ("image") in [127:126]
  i32x8 g1;
  g1[0] = (int)(1u << 16);                                      // data_size=2B
  g1[1] = (int)((tensor_k & 0xffffu) << 16);                    // dim0[15:0]
  g1[2] = (int)(((tensor_k >> 16) & 0xffffu) |
                ((tensor_rows & 0xffffu) << 16));               // dim0[31:16], dim1[15:0]
  g1[3] = (int)(((tensor_rows >> 16) & 0xffffu) |
                ((tile_k & 0xffffu) << 16));                    // dim1[31:16], tile_dim0
  g1[4] = (int)(tile_rows & 0xffffu);                           // tile_dim1 (tile_dim2=0)
  g1[5] = (int)stride_k;                                        // dim0_stride[31:0]
  g1[6] = (int)((stride_k >> 31 >> 1) & 0xffffu);               // dim0_stride[47:32]
  g1[7] = 0;
  i32x4 gz = {0, 0, 0, 0};
#if defined(__clang_major__) && (__clang_major__ >= 23)
  i32x8 gz8 = {0, 0, 0, 0, 0, 0, 0, 0};
  __builtin_amdgcn_tensor_load_to_lds(g0, g1, gz, gz, gz8, 0);
#else
  __builtin_amdgcn_tensor_load_to_lds(g0, g1, gz, gz, 0);
#endif
}
#endif  // GAE_USE_TDM

// ---------- WMMA helpers ----------
__device__ __forceinline__ v8f wmma_bf16(v16bf a, v16bf b, v8f c) {
  return __builtin_amdgcn_wmma_f32_16x16x32_bf16(
      false, a, false, b, (short)0, c, false, false);
}

// A fragment: 16x32 bf16 row-major, leading dim ld (ISA 7.12.2 16-bit A layout)
__device__ __forceinline__ v16bf load_a_frag(const __bf16* __restrict__ base,
                                             int ld, int k0, int lane) {
  const int row = lane & 15;
  const int ko  = (lane & 16) ? 8 : 0;
  const __bf16* p = base + (size_t)row * ld + k0 + ko;
  FragU u;
  u.h[0] = *(const v8bf*)(p);
  u.h[1] = *(const v8bf*)(p + 16);
  return u.v;
}

// B fragment from column-major B given as Bt[N][K] row-major (leading dim ld):
// lane n holds column n; lanes 0-15 K k0..k0+7,k0+16..k0+23, lanes 16-31 K k0+8..k0+15,k0+24..k0+31.
__device__ __forceinline__ v16bf load_b_frag(const __bf16* __restrict__ baseT,
                                             int ld, int k0, int lane) {
  const int col = lane & 15;
  const int ko  = (lane & 16) ? 8 : 0;
  const __bf16* p = baseT + (size_t)col * ld + k0 + ko;
  FragU u;
  u.h[0] = *(const v8bf*)(p);
  u.h[1] = *(const v8bf*)(p + 16);
  return u.v;
}

// ---------- conversion kernels ----------
// Vectorized fp32->bf16: 2x b128 loads + 1x b128 store per 8 elements.
__global__ void gae_cvt_f32_bf16_v8(const float* __restrict__ in,
                                    __bf16* __restrict__ out, size_t n8) {
  size_t i = (size_t)blockIdx.x * blockDim.x + threadIdx.x;
  const size_t stride = (size_t)gridDim.x * blockDim.x;
  for (; i < n8; i += stride) {
    v4f x0 = *(const v4f*)(in + i * 8);
    v4f x1 = *(const v4f*)(in + i * 8 + 4);
    v8bf o;
#pragma unroll
    for (int j = 0; j < 4; ++j) {
      o[j]     = (__bf16)x0[j];
      o[j + 4] = (__bf16)x1[j];
    }
    *(v8bf*)(out + i * 8) = o;
  }
}

// in[R][C] fp32 -> out[C][R] bf16 (pre-transpose small weight matrices)
__global__ void gae_cvt_transpose(const float* __restrict__ in,
                                  __bf16* __restrict__ out, int R, int C) {
  int idx = blockIdx.x * blockDim.x + threadIdx.x;
  if (idx < R * C) {
    int r = idx / C, c = idx % C;
    out[(size_t)c * R + r] = (__bf16)in[idx];
  }
}

// ---------- generic GEMM: C[M,Nc] = act(A[M,K] * B[K,Nc]), bf16 in/out, f32 acc
// A row-major (ld=K), B given transposed Bt[Nc][K] (ld=K).
// TSTORE=false: C row-major [M][Nc]; TSTORE=true: C stored transposed [Nc][M].
// block = 256 threads (8 waves); wave w -> rows [by*128+16w,+16), cols [bx*64,+64).
// The 64xGAE_KC B panel is double-buffered in LDS via TDM: wave 0 issues the
// DMA for chunk i+1, then s_wait_tensorcnt(1) guarantees chunk i has landed
// (same-wave TDM ops complete in order), overlapping DMA with WMMA compute.
template <bool RELU, bool TSTORE>
__global__ void gae_gemm_bt(const __bf16* __restrict__ A,
                            const __bf16* __restrict__ Bt,
                            __bf16* __restrict__ C,
                            int M, int Nc, int K) {
#if GAE_USE_TDM
  __shared__ __align__(16) __bf16 Bs[2][64][GAE_KC];
#else
  __shared__ __align__(16) __bf16 Bs[1][64][GAE_KC];
#endif

  const int lane = threadIdx.x & 31;
  const int wave = threadIdx.x >> 5;
  const int row0 = blockIdx.y * 128 + wave * 16;
  const int col0 = blockIdx.x * 64;

  const __bf16* Abase  = A  + (size_t)row0 * K;
  const __bf16* Bpanel = Bt + (size_t)col0 * K;

  v8f acc0 = {}, acc1 = {}, acc2 = {}, acc3 = {};

  const int nkb = K / GAE_KC;

#if GAE_USE_TDM
  if (threadIdx.x == 0) {  // prologue: DMA chunk 0 into buffer 0
    gae_tdm_load_2d(Bpanel, (unsigned)(size_t)&Bs[0][0][0],
                    GAE_KC, 64u, (unsigned)K, (unsigned)Nc, (unsigned)K);
  }
#endif

  for (int ib = 0; ib < nkb; ++ib) {
    const int kb = ib * GAE_KC;

#if GAE_USE_TDM
    if (threadIdx.x == 0) {
      if (ib + 1 < nkb) {  // DMA next chunk into the other buffer
        gae_tdm_load_2d(Bpanel + kb + GAE_KC,
                        (unsigned)(size_t)&Bs[(ib + 1) & 1][0][0],
                        GAE_KC, 64u, (unsigned)K, (unsigned)Nc, (unsigned)K);
        __builtin_amdgcn_s_wait_tensorcnt(1);  // current chunk landed
      } else {
        __builtin_amdgcn_s_wait_tensorcnt(0);  // last chunk landed
      }
    }
    const __bf16* Bsb = &Bs[ib & 1][0][0];
#else
    for (int c = threadIdx.x; c < 64 * GAE_KC / 8; c += 256) {
      int rr = c / (GAE_KC / 8), cc = (c % (GAE_KC / 8)) * 8;
      *(v8bf*)&Bs[0][rr][cc] = *(const v8bf*)(Bpanel + (size_t)rr * K + kb + cc);
    }
    const __bf16* Bsb = &Bs[0][0][0];
#endif
    __syncthreads();  // publish current buffer to all waves

    if (kb + GAE_KC < K) {  // uniform; speculative prefetch of next A chunk
      __builtin_prefetch(Abase + (size_t)(lane & 15) * K + kb + GAE_KC, 0, 0);
    }

#pragma unroll
    for (int k0 = 0; k0 < GAE_KC; k0 += 32) {
      v16bf a = load_a_frag(Abase, K, kb + k0, lane);
      acc0 = wmma_bf16(a, load_b_frag(Bsb,                GAE_KC, k0, lane), acc0);
      acc1 = wmma_bf16(a, load_b_frag(Bsb + 16 * GAE_KC,  GAE_KC, k0, lane), acc1);
      acc2 = wmma_bf16(a, load_b_frag(Bsb + 32 * GAE_KC,  GAE_KC, k0, lane), acc2);
      acc3 = wmma_bf16(a, load_b_frag(Bsb + 48 * GAE_KC,  GAE_KC, k0, lane), acc3);
    }
    __syncthreads();  // all reads done before this buffer is re-filled
  }

  // C fragment layout: lane&15 = column, lanes>=16 own rows 8..15, vgpr v = row
  const int n     = lane & 15;
  const int rbase = (lane & 16) ? 8 : 0;
  v8f accs[4] = {acc0, acc1, acc2, acc3};
#pragma unroll
  for (int t = 0; t < 4; ++t) {
#pragma unroll
    for (int v = 0; v < 8; ++v) {
      float x = accs[t][v];
      if (RELU) x = fmaxf(x, 0.0f);
      const int r = row0 + rbase + v;
      const int c = col0 + t * 16 + n;
      if (TSTORE) C[(size_t)c * M + r] = (__bf16)x;
      else        C[(size_t)r * Nc + c] = (__bf16)x;
    }
  }
}

// ---------- fused scores + row-softmax ----------
// out[r,:] = softmax_c( emb[r,:] . emb[c,:] ) + 1e-10, fp32 output.
// emb: [N][H] bf16 row-major, H=64. One block = 16 rows; 8 waves partition the
// column tiles; two passes (online stats, then write), no N x N intermediate.
__global__ void gae_softmax_scores(const __bf16* __restrict__ emb,
                                   float* __restrict__ out, int N, int H) {
  const int lane = threadIdx.x & 31;
  const int wave = threadIdx.x >> 5;
  const int row0 = blockIdx.x * 16;

  __shared__ float lm[8][16];
  __shared__ float ls[8][16];

  const __bf16* Abase = emb + (size_t)row0 * H;
  const v16bf a0 = load_a_frag(Abase, H, 0,  lane);
  const v16bf a1 = load_a_frag(Abase, H, 32, lane);

  float mx[8], sm[8];
#pragma unroll
  for (int v = 0; v < 8; ++v) { mx[v] = -3.0e38f; sm[v] = 0.0f; }

  const int nt = N / 16;

  for (int t = wave; t < nt; t += 8) {
    const __bf16* bb = emb + (size_t)t * 16 * H;
    v8f acc = {};
    acc = wmma_bf16(a0, load_b_frag(bb, H, 0,  lane), acc);
    acc = wmma_bf16(a1, load_b_frag(bb, H, 32, lane), acc);
#pragma unroll
    for (int v = 0; v < 8; ++v) {
      float x  = acc[v];
      float mn = fmaxf(mx[v], x);
      sm[v] = sm[v] * __expf(mx[v] - mn) + __expf(x - mn);
      mx[v] = mn;
    }
  }

  // reduce across the 16 lanes sharing each row half (xor 1,2,4,8)
#pragma unroll
  for (int v = 0; v < 8; ++v) {
    for (int d = 1; d < 16; d <<= 1) {
      float mo = __shfl_xor(mx[v], d, 32);
      float so = __shfl_xor(sm[v], d, 32);
      float mn = fmaxf(mx[v], mo);
      sm[v] = sm[v] * __expf(mx[v] - mn) + so * __expf(mo - mn);
      mx[v] = mn;
    }
  }

  const int rbase = (lane & 16) ? 8 : 0;
  if ((lane & 15) == 0) {
#pragma unroll
    for (int v = 0; v < 8; ++v) { lm[wave][rbase + v] = mx[v]; ls[wave][rbase + v] = sm[v]; }
  }
  __syncthreads();

  float M_[8], Sinv[8];
#pragma unroll
  for (int v = 0; v < 8; ++v) {
    float m = lm[0][rbase + v], s = ls[0][rbase + v];
    for (int w = 1; w < 8; ++w) {
      float mo = lm[w][rbase + v], so = ls[w][rbase + v];
      float mn = fmaxf(m, mo);
      s = s * __expf(m - mn) + so * __expf(mo - mn);
      m = mn;
    }
    M_[v]   = m;
    Sinv[v] = 1.0f / s;
  }

  for (int t = wave; t < nt; t += 8) {
    const __bf16* bb = emb + (size_t)t * 16 * H;
    v8f acc = {};
    acc = wmma_bf16(a0, load_b_frag(bb, H, 0,  lane), acc);
    acc = wmma_bf16(a1, load_b_frag(bb, H, 32, lane), acc);
    const int col = t * 16 + (lane & 15);
#pragma unroll
    for (int v = 0; v < 8; ++v) {
      out[(size_t)(row0 + rbase + v) * N + col] =
          __expf(acc[v] - M_[v]) * Sinv[v] + 1e-10f;
    }
  }
}

// ---------- launch ----------
extern "C" void kernel_launch(void* const* d_in, const int* in_sizes, int n_in,
                              void* d_out, int out_size, void* d_ws, size_t ws_size,
                              hipStream_t stream) {
  const int N = 8192, D = 1024, H1 = 256, H2 = 64;
  const float* L  = (const float*)d_in[0];
  const float* X  = (const float*)d_in[1];
  const float* W1 = (const float*)d_in[2];
  const float* W2 = (const float*)d_in[3];
  float* out = (float*)d_out;

  char*  ws  = (char*)d_ws;
  size_t off = 0;
  auto carve = [&](size_t bytes) -> __bf16* {
    char* p = ws + off;
    off += (bytes + 255) & ~(size_t)255;
    return (__bf16*)p;
  };
  __bf16* Lb  = carve((size_t)N * N  * 2);   // 128 MB
  __bf16* Xb  = carve((size_t)N * D  * 2);   //  16 MB
  __bf16* W1t = carve((size_t)D * H1 * 2);   // [H1][D]
  __bf16* W2t = carve((size_t)H1 * H2 * 2);  // [H2][H1]
  __bf16* T1t = carve((size_t)N * H1 * 2);   // [H1][N]  (X@W1)^T
  __bf16* Hb  = carve((size_t)N * H1 * 2);   // [N][H1]  relu(L@T1)
  __bf16* T2t = carve((size_t)N * H2 * 2);   // [H2][N]  (H@W2)^T
  __bf16* Eb  = carve((size_t)N * H2 * 2);   // [N][H2]  emb
  (void)ws_size; (void)in_sizes; (void)n_in; (void)out_size;

  gae_cvt_f32_bf16_v8<<<4096, 256, 0, stream>>>(L, Lb, (size_t)N * N / 8);
  gae_cvt_f32_bf16_v8<<<1024, 256, 0, stream>>>(X, Xb, (size_t)N * D / 8);
  gae_cvt_transpose<<<(D * H1 + 255) / 256, 256, 0, stream>>>(W1, W1t, D, H1);
  gae_cvt_transpose<<<(H1 * H2 + 255) / 256, 256, 0, stream>>>(W2, W2t, H1, H2);

  // T1t = (X @ W1)^T           [H1][N]
  gae_gemm_bt<false, true ><<<dim3(H1 / 64, N / 128), 256, 0, stream>>>(Xb, W1t, T1t, N, H1, D);
  // H = relu(L @ T1)           [N][H1]
  gae_gemm_bt<true,  false><<<dim3(H1 / 64, N / 128), 256, 0, stream>>>(Lb, T1t, Hb, N, H1, N);
  // T2t = (H @ W2)^T           [H2][N]
  gae_gemm_bt<false, true ><<<dim3(H2 / 64, N / 128), 256, 0, stream>>>(Hb, W2t, T2t, N, H2, H1);
  // emb = L @ T2               [N][H2]
  gae_gemm_bt<false, false><<<dim3(H2 / 64, N / 128), 256, 0, stream>>>(Lb, T2t, Eb, N, H2, N);
  // out = softmax(emb emb^T) + 1e-10, fused, single write of the 256MB result
  gae_softmax_scores<<<N / 16, 256, 0, stream>>>(Eb, out, N, H2);
}
